// DepthGuidedFeatureVolume_20461224198613
// MI455X (gfx1250) — matrix operand order repaired and checked by reading, not compiled, not run on hardware
//
#include <hip/hip_runtime.h>
#include <hip/hip_bf16.h>

typedef __attribute__((ext_vector_type(16))) _Float16 v16h;
typedef __attribute__((ext_vector_type(8)))  _Float16 v8h;
typedef __attribute__((ext_vector_type(2)))  _Float16 v2h;
typedef __attribute__((ext_vector_type(8)))  float    v8f;
typedef __attribute__((ext_vector_type(4)))  float    v4f;

#define RESO   128
#define NVIEW  8
#define CIN    32
#define HH     256
#define WW     256
#define NVOX   (RESO*RESO*RESO)      // 2,097,152 voxels
#define NTILE  (NVOX/16)             // 131,072 wave-tiles
#define LSTR   56                    // f16 region row stride (112B = 28 banks; 16B-aligned;
                                     // +8 rows -> +32 banks: conflict-free)
#define FSTR   20                    // f32 region row stride (80B = 20 banks; 16B-aligned;
                                     // +8 rows -> +32 banks: conflict-free)

__device__ __forceinline__ float relu(float x) {
    return __builtin_amdgcn_fmed3f(x, 0.0f, __builtin_inff());
}

// ---------------------------------------------------------------------------
// Phase B: project every pixel, scatter-add its 32-ch feature into the volume
// ---------------------------------------------------------------------------
__global__ void dgv_scatter(const float* __restrict__ feats,
                            const float* __restrict__ depth,
                            const float* __restrict__ poses,
                            const float* __restrict__ intr,
                            float* __restrict__ vol) {
    int idx = blockIdx.x * blockDim.x + threadIdx.x;   // 0 .. NVIEW*HH*WW-1
    int w = idx & (WW - 1);
    int h = (idx >> 8) & (HH - 1);
    int n = idx >> 16;                                 // uniform per block

    // general 3x3 inverse of K (faithful to jnp.linalg.inv)
    const float* K = intr + n * 9;
    float k00 = K[0], k01 = K[1], k02 = K[2];
    float k10 = K[3], k11 = K[4], k12 = K[5];
    float k20 = K[6], k21 = K[7], k22 = K[8];
    float det = k00 * (k11 * k22 - k12 * k21)
              - k01 * (k10 * k22 - k12 * k20)
              + k02 * (k10 * k21 - k11 * k20);
    float rdet  = 1.0f / det;
    float inv00 = (k11 * k22 - k12 * k21) * rdet;
    float inv11 = (k00 * k22 - k02 * k20) * rdet;
    float inv02 = (k01 * k12 - k02 * k11) * rdet;
    float inv12 = (k02 * k10 - k00 * k12) * rdet;

    float d   = depth[idx];
    // faithful to the reference's (unusual) use of Kinv entries
    float gxn = ((float)w - inv02) * d / inv00;
    float gyn = ((float)h - inv12) * d / inv11;

    const float* P = poses + n * 16;
    float wx = P[0] * gxn + P[1] * gyn + P[2]  * d + P[3];
    float wy = P[4] * gxn + P[5] * gyn + P[6]  * d + P[7];
    float wz = P[8] * gxn + P[9] * gyn + P[10] * d + P[11];

    float dist = fabsf(wz - d);
    // trunc-toward-zero like .astype(int32)
    int vx = (int)(wx * (1.0f / RESO));
    int vy = (int)(wy * (1.0f / RESO));
    int vz = (int)(wz * (1.0f / RESO));

    bool valid = (dist <= 1.0f)
               & (vx >= 0) & (vx < RESO)
               & (vy >= 0) & (vy < RESO)
               & (vz >= 0) & (vz < RESO);
    if (!valid) return;   // reference scatters zeros to slot 0 -> no-op

    long flat = (((long)vx * RESO + vy) * RESO + vz);
    float* dst = vol + flat * CIN;
    const float* f = feats + (long)n * CIN * HH * WW + (long)h * WW + w;
#pragma unroll
    for (int c = 0; c < CIN; ++c)
        __hip_atomic_fetch_add(&dst[c], f[(long)c * HH * WW],
                               __ATOMIC_RELAXED, __HIP_MEMORY_SCOPE_AGENT);
}

// ---------------------------------------------------------------------------
// Phase C: MLP head 32->32->16->8 over all voxels, 16 voxels per wave-tile,
// 4 x v_wmma_f32_16x16x32_f16 per tile (f16 activations, f32 accumulate).
//  - layer1->2 hop: N-tile-interleaved f16 LDS layout -> packed b32 stores,
//    contiguous ds_load_b128 A2 loads; permutation absorbed into B2 fragment
//  - layer2->3 hop: f32 LDS staging -> paired 2addr b32 stores (no cvt on the
//    store side), b128 loads + cvt_pk on the load side
// ---------------------------------------------------------------------------
__global__ void dgv_mlp(const float* __restrict__ vol,
                        const float* __restrict__ W1, const float* __restrict__ b1,
                        const float* __restrict__ W2, const float* __restrict__ b2,
                        const float* __restrict__ W3, const float* __restrict__ b3,
                        float* __restrict__ out) {
    __shared__ _Float16 ldsA[8][16][LSTR];   // layer1 -> layer2 (f16, interleaved)
    __shared__ float    ldsB[8][16][FSTR];   // layer2 -> layer3 (f32)

    const int lane = threadIdx.x & 31;
    const int wid  = threadIdx.x >> 5;
    const int col  = lane & 15;        // N index for B / C / D fragments
    const int hi   = lane >> 4;
    const int kb   = hi * 16;          // unused; B fragments also use klo (lanes 16-31: K=8..15 / 24..31)
    const int klo  = hi * 8;           // A-fragment K base  (lanes 16-31: K=8..15 / 24..31)

    // ---- loop-invariant weight fragments (B is K x N, i.e. W^T) ----
    v16h B1a, B1b, B2f, B3f;
#pragma unroll
    for (int i = 0; i < 16; ++i) {
        int k = (i < 8) ? klo + i : 8 + klo + i;
        B1a[i] = (_Float16)W1[col * 32 + k];               // N = 0..15
        B1b[i] = (_Float16)W1[(col + 16) * 32 + k];        // N = 16..31
        // h1 channel stored at LDS position p is ch(p) = (p&1)*16 + (p>>1);
        // B2 row k must supply W2[n][ch(k)]
        int ch = ((k & 1) << 4) | (k >> 1);
        B2f[i] = (_Float16)W2[col * 32 + ch];              // 16x32 (permuted K)
        B3f[i] = (col < 8 && k < 16) ? (_Float16)W3[col * 16 + k] : (_Float16)0.0f;
    }
    const float bias1a = b1[col];
    const float bias1b = b1[col + 16];
    const float bias2  = b2[col];
    const float bias3  = (col < 8) ? b3[col] : 0.0f;

    const int wavesTotal = gridDim.x * (blockDim.x >> 5);
    const int wave0      = blockIdx.x * (blockDim.x >> 5) + wid;
    const int row        = col;        // A-fragment row = lane & 15

    for (int t = wave0; t < NTILE; t += wavesTotal) {   // wave-uniform: EXEC all-1s at WMMAs
        const long vb = (long)t * 16;

        // ---- A1: 16 voxels x 32 channels, f32 -> f16, ISA A-layout ----
        const float* vp = vol + (vb + row) * CIN;
        v16h A1;
#pragma unroll
        for (int i = 0; i < 8; ++i) A1[i]     = (_Float16)vp[klo + i];
#pragma unroll
        for (int i = 0; i < 8; ++i) A1[8 + i] = (_Float16)vp[16 + klo + i];

        // speculative prefetch of next tile's rows (dropped silently if unmapped)
        __builtin_prefetch(vol + (vb + (long)wavesTotal * 16 + row) * CIN, 0, 1);

        // ---- layer 1: (16x32) @ W1^T -> 16x32, bias preloaded into C ----
        v8f c1a, c1b;
#pragma unroll
        for (int r = 0; r < 8; ++r) { c1a[r] = bias1a; c1b[r] = bias1b; }
        c1a = __builtin_amdgcn_wmma_f32_16x16x32_f16(false, A1, false, B1a, (short)0, c1a, false, false);
        c1b = __builtin_amdgcn_wmma_f32_16x16x32_f16(false, A1, false, B1b, (short)0, c1b, false, false);

        // ReLU + D->A transpose: packed b32 stores, interleaved channel layout
#pragma unroll
        for (int r = 0; r < 8; ++r) {
            int m = r + 8 * hi;
            v2h pv;
            pv[0] = (_Float16)relu(c1a[r]);    // channel col      -> position 2*col
            pv[1] = (_Float16)relu(c1b[r]);    // channel col+16   -> position 2*col+1
            *reinterpret_cast<v2h*>(&ldsA[wid][m][2 * col]) = pv;
        }
        // A2: two contiguous 16B LDS vector loads per lane (positions = K indices)
        v8h a2lo = *reinterpret_cast<const v8h*>(&ldsA[wid][row][klo]);
        v8h a2hi = *reinterpret_cast<const v8h*>(&ldsA[wid][row][16 + klo]);
        v16h A2;
#pragma unroll
        for (int i = 0; i < 8; ++i) { A2[i] = a2lo[i]; A2[8 + i] = a2hi[i]; }

        // ---- layer 2: (16x32) @ W2^T(permuted) -> 16x16 ----
        v8f c2;
#pragma unroll
        for (int r = 0; r < 8; ++r) c2[r] = bias2;
        c2 = __builtin_amdgcn_wmma_f32_16x16x32_f16(false, A2, false, B2f, (short)0, c2, false, false);

        // ReLU + D->A transpose in f32 (no cvt on store side; pairs -> 2addr b32)
#pragma unroll
        for (int r = 0; r < 8; ++r)
            ldsB[wid][r + 8 * hi][col] = relu(c2[r]);

        // ---- layer 3: (16x16, K padded to 32) @ W3^T -> 16x8 ----
        v4f a3f0 = *reinterpret_cast<const v4f*>(&ldsB[wid][row][klo]);
        v4f a3f1 = *reinterpret_cast<const v4f*>(&ldsB[wid][row][klo + 4]);
        v16h A3;
#pragma unroll
        for (int i = 0; i < 4; ++i) { A3[i] = (_Float16)a3f0[i]; A3[4 + i] = (_Float16)a3f1[i]; }
#pragma unroll
        for (int i = 8; i < 16; ++i) A3[i] = (_Float16)0.0f;  // K >= 16 padding
        v8f c3;
#pragma unroll
        for (int r = 0; r < 8; ++r) c3[r] = bias3;
        c3 = __builtin_amdgcn_wmma_f32_16x16x32_f16(false, A3, false, B3f, (short)0, c3, false, false);

        // ---- store 16 voxels x 8 outputs (lanes with N>=8 idle) ----
        if (col < 8) {
            float* op = out + (vb + 8 * hi) * 8 + col;
#pragma unroll
            for (int r = 0; r < 8; ++r) op[r * 8] = c3[r];
        }
    }
}

// ---------------------------------------------------------------------------
extern "C" void kernel_launch(void* const* d_in, const int* in_sizes, int n_in,
                              void* d_out, int out_size, void* d_ws, size_t ws_size,
                              hipStream_t stream) {
    const float* feats = (const float*)d_in[0];
    const float* depth = (const float*)d_in[1];
    const float* poses = (const float*)d_in[2];
    const float* intr  = (const float*)d_in[3];
    const float* W1    = (const float*)d_in[4];
    const float* b1    = (const float*)d_in[5];
    const float* W2    = (const float*)d_in[6];
    const float* b2    = (const float*)d_in[7];
    const float* W3    = (const float*)d_in[8];
    const float* b3    = (const float*)d_in[9];
    float* out = (float*)d_out;
    float* vol = (float*)d_ws;                       // NVOX * 32 f32 = 256 MiB scratch

    hipMemsetAsync(vol, 0, (size_t)NVOX * CIN * sizeof(float), stream);

    int npix = NVIEW * HH * WW;                      // 524,288
    dgv_scatter<<<npix / 256, 256, 0, stream>>>(feats, depth, poses, intr, vol);

    dgv_mlp<<<1024, 256, 0, stream>>>(vol, W1, b1, W2, b2, W3, b3, out);
}
